// NysAttention_58411555225987
// MI455X (gfx1250) — compile-verified, matrix operand order checked by reading, not run on hardware
//
#include <hip/hip_runtime.h>
#include <hip/hip_bf16.h>

// ---------------------------------------------------------------------------
// Nystrom attention for MI455X (gfx1250, wave32, WMMA).
// Big GEMMs: v_wmma_f32_16x16x32_bf16, B tile staged in LDS (96KB),
// 32x64 output per wave (2 A frags share each B frag -> half the LDS reads),
// software-pipelined A fragments.
// Newton-Schulz 64x64 pseudo-inverse: v_wmma_f32_16x16x4_f32 in LDS.
// ---------------------------------------------------------------------------

#define B_ 8
#define N_ 4096
#define C_ 768
#define H_ 12
#define D_ 64
#define L_ 64
#define BH_ (B_ * H_)

typedef __bf16 bf16;
typedef __attribute__((ext_vector_type(16))) __bf16 v16bf;
typedef __attribute__((ext_vector_type(8)))  __bf16 v8bf;
typedef __attribute__((ext_vector_type(8)))  float  v8f;
typedef __attribute__((ext_vector_type(2)))  float  v2f;

// ---- WMMA helpers ---------------------------------------------------------

__device__ __forceinline__ v8f wmma_bf16(v16bf a, v16bf b, v8f c) {
    return __builtin_amdgcn_wmma_f32_16x16x32_bf16(false, a, false, b,
                                                   (short)0, c, false, false);
}

__device__ __forceinline__ v8f wmma_f32(v2f a, v2f b, v8f c) {
    return __builtin_amdgcn_wmma_f32_16x16x4_f32(false, a, false, b,
                                                 (short)0, c, false, false);
}

// bf16 A/B fragment: lane -> row (A) / col (B); halves j<8 at k=kb+j,
// j>=8 at k=kb+16+(j-8); kb = (lane>>4)*8. Two contiguous 16B loads.
__device__ __forceinline__ v16bf load_frag_b(const bf16* rowptr, int kb) {
    v8bf lo = *reinterpret_cast<const v8bf*>(rowptr + kb);
    v8bf hi = *reinterpret_cast<const v8bf*>(rowptr + kb + 16);
    v16bf r;
#pragma unroll
    for (int j = 0; j < 8; ++j) { r[j] = lo[j]; r[j + 8] = hi[j]; }
    return r;
}

// ---- conversion kernels ---------------------------------------------------

__global__ void cvt_f32_bf16(const float* __restrict__ src, bf16* __restrict__ dst, int n) {
    for (int i = blockIdx.x * blockDim.x + threadIdx.x; i < n; i += gridDim.x * blockDim.x)
        dst[i] = (bf16)src[i];
}

// src [K][Ncols] row-major -> dst [Ncols][K] (column-major of src), bf16
__global__ void transpose_cvt(const float* __restrict__ src, bf16* __restrict__ dst,
                              int K, int Ncols) {
    int i = blockIdx.x * blockDim.x + threadIdx.x;
    if (i < K * Ncols) {
        int k = i / Ncols, c = i % Ncols;
        dst[c * K + k] = (bf16)src[i];
    }
}

// ---- GEMM 1: qkv = x @ w_qkv, scattered into Q (scaled), K, V --------------
// grid (36, 128); block 256 (8 waves). Block: 256 rows x 64 cols, K=768.
// Wave: 32 rows x 64 cols. B tile (64x768) staged once in LDS.

__global__ __launch_bounds__(256)
void gemm_qkv(const bf16* __restrict__ xbf, const bf16* __restrict__ wT,
              bf16* __restrict__ Qbf, bf16* __restrict__ Kbf, float* __restrict__ Vf) {
    __shared__ __align__(16) bf16 Bs[64 * C_];          // 96 KB
    const int tid = threadIdx.x;
    const int wave = tid >> 5, lane = tid & 31;
    const int lr = lane & 15, kb = (lane >> 4) * 8, hi8 = (lane >> 4) << 3;
    const int coltile = blockIdx.x;                     // 0..35
    const int rowbase = blockIdx.y * 256 + wave * 32;
    const int colbase = coltile * 64;

    // cooperative stage of B tile: contiguous 96KB region of wT
    {
        const v8bf* src = reinterpret_cast<const v8bf*>(wT + (size_t)colbase * C_);
        v8bf* dst = reinterpret_cast<v8bf*>(Bs);
#pragma unroll 4
        for (int i = tid; i < 64 * C_ / 8; i += 256) dst[i] = src[i];
    }
    __syncthreads();

    const bf16* arow0 = xbf + (size_t)(rowbase + lr) * C_;
    const bf16* arow1 = arow0 + 16 * C_;
    v8f acc[2][4] = {{{}, {}, {}, {}}, {{}, {}, {}, {}}};
    v16bf a0 = load_frag_b(arow0, kb);
    v16bf a1 = load_frag_b(arow1, kb);
    for (int kk = 0; kk < C_; kk += 32) {
        v16bf a0n = a0, a1n = a1;
        if (kk + 32 < C_) {                              // overlap with WMMAs
            a0n = load_frag_b(arow0 + kk + 32, kb);
            a1n = load_frag_b(arow1 + kk + 32, kb);
        }
#pragma unroll
        for (int s = 0; s < 4; ++s) {
            v16bf b = load_frag_b(&Bs[(s * 16 + lr) * C_ + kk], kb);
            acc[0][s] = wmma_bf16(a0, b, acc[0][s]);
            acc[1][s] = wmma_bf16(a1, b, acc[1][s]);
        }
        a0 = a0n; a1 = a1n;
    }

    const int which = coltile / H_;                     // 0=Q 1=K 2=V
    const int h = coltile % H_;
#pragma unroll
    for (int rr = 0; rr < 2; ++rr) {
#pragma unroll
        for (int s = 0; s < 4; ++s) {
#pragma unroll
            for (int r = 0; r < 8; ++r) {
                int m = rowbase + rr * 16 + r + hi8;
                int b = m >> 12, n = m & (N_ - 1);
                int d = s * 16 + lr;
                size_t idx = (((size_t)b * H_ + h) * N_ + n) * D_ + d;
                float v = acc[rr][s][r];
                if (which == 0)      Qbf[idx] = (bf16)(v * 0.125f);  // 1/sqrt(64)
                else if (which == 1) Kbf[idx] = (bf16)v;
                else                 Vf[idx]  = v;
            }
        }
    }
}

// ---- landmark means -------------------------------------------------------
// grid BH_*L_, block 64. landmark l = mean of rows [l*64, l*64+64)

__global__ void landmarks(const bf16* __restrict__ Qbf, const bf16* __restrict__ Kbf,
                          bf16* __restrict__ Qlm, bf16* __restrict__ Klm) {
    int bh = blockIdx.x >> 6, l = blockIdx.x & 63, d = threadIdx.x;
    const bf16* q = Qbf + ((size_t)bh * N_ + l * 64) * D_ + d;
    const bf16* k = Kbf + ((size_t)bh * N_ + l * 64) * D_ + d;
    float sq = 0.f, sk = 0.f;
    for (int s = 0; s < 64; ++s) { sq += (float)q[s * D_]; sk += (float)k[s * D_]; }
    Qlm[(bh * L_ + l) * D_ + d] = (bf16)(sq * (1.f / 64.f));
    Klm[(bh * L_ + l) * D_ + d] = (bf16)(sk * (1.f / 64.f));
}

// ---- 64x64 f32 matmul in LDS: dst = diag*I + scale*(A@B) ------------------

__device__ void mm64(float* __restrict__ dst, const float* __restrict__ A,
                     const float* __restrict__ Bm, float scale, float diag, int tid) {
    const int wave = tid >> 5, lane = tid & 31;
    const int lr = lane & 15, kb2 = (lane >> 4) * 2, hi8 = (lane >> 4) << 3;
#pragma unroll
    for (int i = 0; i < 2; ++i) {
        int t = wave * 2 + i, tr = t >> 2, tc = t & 3;
        v8f acc = {};
        for (int kk = 0; kk < 64; kk += 4) {
            v2f a, b;
            a.x = A[(tr * 16 + lr) * 64 + kk + kb2];
            a.y = A[(tr * 16 + lr) * 64 + kk + kb2 + 1];
            b.x = Bm[(kk + kb2) * 64 + tc * 16 + lr];
            b.y = Bm[(kk + kb2 + 1) * 64 + tc * 16 + lr];
            acc = wmma_f32(a, b, acc);
        }
#pragma unroll
        for (int r = 0; r < 8; ++r) {
            int row = tr * 16 + r + hi8, col = tc * 16 + lr;
            float v = acc[r] * scale;
            if (row == col) v += diag;
            dst[row * 64 + col] = v;
        }
    }
    __syncthreads();
}

// ---- kernel2 softmax + Newton-Schulz pseudo-inverse -----------------------
// grid BH_, block 256

__global__ __launch_bounds__(256)
void k2inv_kernel(const bf16* __restrict__ Qlm, const bf16* __restrict__ Klm,
                  bf16* __restrict__ Kinv) {
    __shared__ __align__(16) float K2[4096];
    __shared__ __align__(16) float KV[4096];
    __shared__ __align__(16) float Am[4096];
    __shared__ __align__(16) float BufA[4096];
    __shared__ __align__(16) float BufB[4096];
    __shared__ float red[64];
    __shared__ float denom_s;

    const int bh = blockIdx.x, tid = threadIdx.x;
    const int wave = tid >> 5, lane = tid & 31;
    const int lr = lane & 15, kb = (lane >> 4) * 8, hi8 = (lane >> 4) << 3;
    const bf16* qp = Qlm + (size_t)bh * L_ * D_;
    const bf16* kp = Klm + (size_t)bh * L_ * D_;

    // scores = Qlm @ Klm^T  (bf16 WMMA, 16 tiles / 8 waves)
#pragma unroll
    for (int i = 0; i < 2; ++i) {
        int t = wave * 2 + i, tr = t >> 2, tc = t & 3;
        v8f acc = {};
        for (int kk = 0; kk < D_; kk += 32) {
            v16bf a = load_frag_b(qp + (tr * 16 + lr) * D_ + kk, kb);
            v16bf b = load_frag_b(kp + (tc * 16 + lr) * D_ + kk, kb);
            acc = wmma_bf16(a, b, acc);
        }
#pragma unroll
        for (int r = 0; r < 8; ++r)
            K2[(tr * 16 + r + hi8) * 64 + tc * 16 + lr] = acc[r];
    }
    __syncthreads();

    // row softmax
    if (tid < 64) {
        float m = -1e30f;
        for (int j = 0; j < 64; ++j) m = fmaxf(m, K2[tid * 64 + j]);
        float s = 0.f;
        for (int j = 0; j < 64; ++j) { float e = __expf(K2[tid * 64 + j] - m); s += e; K2[tid * 64 + j] = e; }
        float inv = 1.f / s;
        for (int j = 0; j < 64; ++j) K2[tid * 64 + j] *= inv;
    }
    __syncthreads();

    // denom = max_col(sum_rows(K2))
    if (tid < 64) {
        float s = 0.f;
        for (int r = 0; r < 64; ++r) s += K2[r * 64 + tid];
        red[tid] = s;
    }
    __syncthreads();
    if (tid == 0) {
        float m = red[0];
        for (int j = 1; j < 64; ++j) m = fmaxf(m, red[j]);
        denom_s = m;
    }
    __syncthreads();
    float inv_d = 1.f / denom_s;

    // V0 = K2^T / denom
    for (int i = tid; i < 4096; i += 256) {
        int l = i >> 6, m = i & 63;
        BufA[i] = K2[m * 64 + l] * inv_d;
    }
    __syncthreads();

    float* pV = BufA;
    float* pN = BufB;
    for (int it = 0; it < 6; ++it) {
        mm64(KV, K2, pV, 1.f, 0.f, tid);        // KV = K2 @ V
        // Am = 7I - KV
        for (int i = tid; i < 4096; i += 256) {
            int l = i >> 6, m = i & 63;
            Am[i] = (l == m ? 7.f : 0.f) - KV[i];
        }
        __syncthreads();
        mm64(pN, KV, Am, -1.f, 15.f, tid);      // pN = 15I - KV@Am
        mm64(Am, KV, pN, -1.f, 13.f, tid);      // Am = 13I - KV@pN
        mm64(pN, pV, Am, 0.25f, 0.f, tid);      // Vnew = 0.25 * V @ Am
        float* tmp = pV; pV = pN; pN = tmp;
    }

    for (int i = tid; i < 4096; i += 256)
        Kinv[(size_t)bh * 4096 + i] = (bf16)pV[i];
}

// ---- kernel3 streaming softmax + (kernel3@V), then M = K2inv @ R ----------
// grid BH_, block 256. Two passes over N (rowmax, then exp/accumulate).

__global__ __launch_bounds__(256)
void k3v_kernel(const bf16* __restrict__ Qlm, const bf16* __restrict__ Kbf,
                const float* __restrict__ Vf, const bf16* __restrict__ Kinv,
                bf16* __restrict__ MT) {
    __shared__ __align__(16) float S[4096];
    __shared__ __align__(16) bf16  Pb[4096];
    __shared__ __align__(16) bf16  VT[4096];
    __shared__ __align__(16) bf16  RT[4096];
    __shared__ float rowmax[64];
    __shared__ float rowsum[64];

    const int bh = blockIdx.x, tid = threadIdx.x;
    const int wave = tid >> 5, lane = tid & 31;
    const int lr = lane & 15, kb = (lane >> 4) * 8, hi8 = (lane >> 4) << 3;
    const bf16*  qlm = Qlm + (size_t)bh * L_ * D_;
    const bf16*  kp  = Kbf + (size_t)bh * N_ * D_;
    const float* vp  = Vf  + (size_t)bh * N_ * D_;

    if (tid < 64) { rowmax[tid] = -1e30f; rowsum[tid] = 0.f; }
    __syncthreads();

    // pass 1: row maxima of Qlm @ K^T over all N
    for (int ch = 0; ch < N_ / 64; ++ch) {
#pragma unroll
        for (int i = 0; i < 2; ++i) {
            int t = wave * 2 + i, tr = t >> 2, tc = t & 3;
            v8f acc = {};
            for (int kk = 0; kk < D_; kk += 32) {
                v16bf a = load_frag_b(qlm + (tr * 16 + lr) * D_ + kk, kb);
                v16bf b = load_frag_b(kp + (size_t)(ch * 64 + tc * 16 + lr) * D_ + kk, kb);
                acc = wmma_bf16(a, b, acc);
            }
#pragma unroll
            for (int r = 0; r < 8; ++r)
                S[(tr * 16 + r + hi8) * 64 + tc * 16 + lr] = acc[r];
        }
        __syncthreads();
        if (tid < 64) {
            float m = rowmax[tid];
            for (int j = 0; j < 64; ++j) m = fmaxf(m, S[tid * 64 + j]);
            rowmax[tid] = m;
        }
        __syncthreads();
    }

    // pass 2: accumulate O = sum exp(S)·V in WMMA registers
    v8f accO[2] = {{}, {}};
    for (int ch = 0; ch < N_ / 64; ++ch) {
#pragma unroll
        for (int i = 0; i < 2; ++i) {
            int t = wave * 2 + i, tr = t >> 2, tc = t & 3;
            v8f acc = {};
            for (int kk = 0; kk < D_; kk += 32) {
                v16bf a = load_frag_b(qlm + (tr * 16 + lr) * D_ + kk, kb);
                v16bf b = load_frag_b(kp + (size_t)(ch * 64 + tc * 16 + lr) * D_ + kk, kb);
                acc = wmma_bf16(a, b, acc);
            }
#pragma unroll
            for (int r = 0; r < 8; ++r)
                S[(tr * 16 + r + hi8) * 64 + tc * 16 + lr] = acc[r];
        }
        // stage V chunk transposed: VT[d][key]
        for (int i = tid; i < 4096; i += 256) {
            int k_ = i >> 6, d_ = i & 63;
            VT[d_ * 64 + k_] = (bf16)vp[(size_t)(ch * 64 + k_) * D_ + d_];
        }
        __syncthreads();
        if (tid < 64) {
            float m = rowmax[tid], s = 0.f;
            for (int j = 0; j < 64; ++j) {
                float e = __expf(S[tid * 64 + j] - m);
                s += e;
                Pb[tid * 64 + j] = (bf16)e;
            }
            rowsum[tid] += s;
        }
        __syncthreads();
#pragma unroll
        for (int i = 0; i < 2; ++i) {
            int t = wave * 2 + i, tr = t >> 2, tc = t & 3;
            v8f acc = accO[i];
            for (int kk = 0; kk < 64; kk += 32) {
                v16bf a = load_frag_b(&Pb[(tr * 16 + lr) * 64 + kk], kb);
                v16bf b = load_frag_b(&VT[(tc * 16 + lr) * 64 + kk], kb);
                acc = wmma_bf16(a, b, acc);
            }
            accO[i] = acc;
        }
        __syncthreads();
    }

    // normalize and store transposed R into LDS: RT[d][l] = O[l][d]/rowsum[l]
#pragma unroll
    for (int i = 0; i < 2; ++i) {
        int t = wave * 2 + i, tr = t >> 2, tc = t & 3;
#pragma unroll
        for (int r = 0; r < 8; ++r) {
            int l = tr * 16 + r + hi8, d = tc * 16 + lr;
            RT[d * 64 + l] = (bf16)(accO[i][r] / rowsum[l]);
        }
    }
    __syncthreads();

    // M = K2inv @ R ; store transposed MT[d][l]
    const bf16* kinvp = Kinv + (size_t)bh * 4096;
#pragma unroll
    for (int i = 0; i < 2; ++i) {
        int t = wave * 2 + i, tr = t >> 2, tc = t & 3;
        v8f acc = {};
        for (int kk = 0; kk < 64; kk += 32) {
            v16bf a = load_frag_b(kinvp + (tr * 16 + lr) * 64 + kk, kb);
            v16bf b = load_frag_b(&RT[(tc * 16 + lr) * 64 + kk], kb);
            acc = wmma_bf16(a, b, acc);
        }
#pragma unroll
        for (int r = 0; r < 8; ++r) {
            int l = tr * 16 + r + hi8, d = tc * 16 + lr;
            MT[(size_t)bh * 4096 + d * 64 + l] = (bf16)acc[r];
        }
    }
}

// ---- kernel1 softmax + SV = kernel1 @ M -----------------------------------
// grid (BH_, N_/128), block 256. Wave handles 16 rows.

__global__ __launch_bounds__(256)
void sv_kernel(const bf16* __restrict__ Qbf, const bf16* __restrict__ Klm,
               const bf16* __restrict__ MT, bf16* __restrict__ SVbf) {
    __shared__ __align__(16) float S[128 * 64];
    __shared__ __align__(16) bf16  Pb[128 * 64];

    const int bh = blockIdx.x, b = bh / H_, h = bh % H_;
    const int nbase = blockIdx.y * 128;
    const int tid = threadIdx.x, wave = tid >> 5, lane = tid & 31;
    const int lr = lane & 15, kb = (lane >> 4) * 8, hi8 = (lane >> 4) << 3;

    const bf16* qp  = Qbf + ((size_t)bh * N_ + nbase) * D_;
    const bf16* klm = Klm + (size_t)bh * L_ * D_;

    // scores = Q_chunk @ Klm^T
    {
        v8f acc[4] = {{}, {}, {}, {}};
        for (int kk = 0; kk < D_; kk += 32) {
            v16bf a = load_frag_b(qp + (wave * 16 + lr) * D_ + kk, kb);
#pragma unroll
            for (int tc = 0; tc < 4; ++tc) {
                v16bf bfr = load_frag_b(klm + (tc * 16 + lr) * D_ + kk, kb);
                acc[tc] = wmma_bf16(a, bfr, acc[tc]);
            }
        }
#pragma unroll
        for (int tc = 0; tc < 4; ++tc)
#pragma unroll
            for (int r = 0; r < 8; ++r)
                S[(wave * 16 + r + hi8) * 64 + tc * 16 + lr] = acc[tc][r];
    }
    __syncthreads();

    // row softmax (128 rows)
    if (tid < 128) {
        float m = -1e30f;
        for (int j = 0; j < 64; ++j) m = fmaxf(m, S[tid * 64 + j]);
        float s = 0.f;
        for (int j = 0; j < 64; ++j) { float e = __expf(S[tid * 64 + j] - m); s += e; S[tid * 64 + j] = e; }
        float inv = 1.f / s;
        for (int j = 0; j < 64; ++j) Pb[tid * 64 + j] = (bf16)(S[tid * 64 + j] * inv);
    }
    __syncthreads();

    // SV = P @ M   (B frag = MT rows, contiguous over l)
    const bf16* mt = MT + (size_t)bh * 4096;
    v8f acc[4] = {{}, {}, {}, {}};
    for (int kk = 0; kk < 64; kk += 32) {
        v16bf a = load_frag_b(&Pb[(wave * 16 + lr) * 64 + kk], kb);
#pragma unroll
        for (int tc = 0; tc < 4; ++tc) {
            v16bf bfr = load_frag_b(mt + (tc * 16 + lr) * 64 + kk, kb);
            acc[tc] = wmma_bf16(a, bfr, acc[tc]);
        }
    }
#pragma unroll
    for (int tc = 0; tc < 4; ++tc)
#pragma unroll
        for (int r = 0; r < 8; ++r) {
            int n = nbase + wave * 16 + r + hi8;
            int d = tc * 16 + lr;
            SVbf[((size_t)b * N_ + n) * C_ + h * 64 + d] = (bf16)acc[tc][r];
        }
}

// ---- GEMM 2: out = SV @ w_proj + b_proj + V_perm --------------------------
// grid (12, 128); block 256. col-tile == head h. B tile staged in LDS.
// Wave: 32 rows x 64 cols.

__global__ __launch_bounds__(256)
void gemm_out(const bf16* __restrict__ SVbf, const bf16* __restrict__ wT,
              const float* __restrict__ bproj, const float* __restrict__ Vf,
              float* __restrict__ out) {
    __shared__ __align__(16) bf16 Bs[64 * C_];          // 96 KB
    const int tid = threadIdx.x;
    const int wave = tid >> 5, lane = tid & 31;
    const int lr = lane & 15, kb = (lane >> 4) * 8, hi8 = (lane >> 4) << 3;
    const int h = blockIdx.x;
    const int rowbase = blockIdx.y * 256 + wave * 32;

    {
        const v8bf* src = reinterpret_cast<const v8bf*>(wT + (size_t)(h * 64) * C_);
        v8bf* dst = reinterpret_cast<v8bf*>(Bs);
#pragma unroll 4
        for (int i = tid; i < 64 * C_ / 8; i += 256) dst[i] = src[i];
    }
    __syncthreads();

    const bf16* arow0 = SVbf + (size_t)(rowbase + lr) * C_;
    const bf16* arow1 = arow0 + 16 * C_;
    v8f acc[2][4] = {{{}, {}, {}, {}}, {{}, {}, {}, {}}};
    v16bf a0 = load_frag_b(arow0, kb);
    v16bf a1 = load_frag_b(arow1, kb);
    for (int kk = 0; kk < C_; kk += 32) {
        v16bf a0n = a0, a1n = a1;
        if (kk + 32 < C_) {
            a0n = load_frag_b(arow0 + kk + 32, kb);
            a1n = load_frag_b(arow1 + kk + 32, kb);
        }
#pragma unroll
        for (int s = 0; s < 4; ++s) {
            v16bf b = load_frag_b(&Bs[(s * 16 + lr) * C_ + kk], kb);
            acc[0][s] = wmma_bf16(a0, b, acc[0][s]);
            acc[1][s] = wmma_bf16(a1, b, acc[1][s]);
        }
        a0 = a0n; a1 = a1n;
    }

#pragma unroll
    for (int rr = 0; rr < 2; ++rr)
#pragma unroll
        for (int s = 0; s < 4; ++s)
#pragma unroll
            for (int r = 0; r < 8; ++r) {
                int m = rowbase + rr * 16 + r + hi8;
                int b = m >> 12, n = m & (N_ - 1);
                int d = s * 16 + lr;
                int col = h * 64 + d;
                out[(size_t)m * C_ + col] =
                    acc[rr][s][r] + bproj[col] + Vf[(((size_t)b * H_ + h) * N_ + n) * D_ + d];
            }
}

// ---------------------------------------------------------------------------

extern "C" void kernel_launch(void* const* d_in, const int* in_sizes, int n_in,
                              void* d_out, int out_size, void* d_ws, size_t ws_size,
                              hipStream_t stream) {
    const float* x      = (const float*)d_in[0];
    const float* w_qkv  = (const float*)d_in[1];
    const float* w_proj = (const float*)d_in[2];
    const float* b_proj = (const float*)d_in[3];
    float* out = (float*)d_out;

    char* ws = (char*)d_ws;
    size_t off = 0;
    auto alloc = [&](size_t bytes) -> void* {
        void* p = ws + off;
        off = (off + bytes + 255) & ~(size_t)255;
        return p;
    };

    const size_t BNC  = (size_t)B_ * N_ * C_;        // 25,165,824
    const size_t BHND = (size_t)BH_ * N_ * D_;       // 25,165,824

    bf16*  xbf    = (bf16*)alloc(BNC * 2);
    bf16*  wqkvT  = (bf16*)alloc((size_t)3 * C_ * C_ * 2);
    bf16*  wprojT = (bf16*)alloc((size_t)C_ * C_ * 2);
    bf16*  Qbf    = (bf16*)alloc(BHND * 2);
    bf16*  Kbf    = (bf16*)alloc(BHND * 2);
    float* Vf     = (float*)alloc(BHND * 4);
    bf16*  Qlm    = (bf16*)alloc((size_t)BH_ * 4096 * 2);
    bf16*  Klm    = (bf16*)alloc((size_t)BH_ * 4096 * 2);
    bf16*  Kinv   = (bf16*)alloc((size_t)BH_ * 4096 * 2);
    bf16*  MT     = (bf16*)alloc((size_t)BH_ * 4096 * 2);
    bf16*  SVbf   = (bf16*)alloc(BNC * 2);

    cvt_f32_bf16<<<2048, 256, 0, stream>>>(x, xbf, (int)BNC);
    transpose_cvt<<<(C_ * 3 * C_ + 255) / 256, 256, 0, stream>>>(w_qkv, wqkvT, C_, 3 * C_);
    transpose_cvt<<<(C_ * C_ + 255) / 256, 256, 0, stream>>>(w_proj, wprojT, C_, C_);

    gemm_qkv<<<dim3(36, 128), 256, 0, stream>>>(xbf, wqkvT, Qbf, Kbf, Vf);
    landmarks<<<BH_ * L_, 64, 0, stream>>>(Qbf, Kbf, Qlm, Klm);
    k2inv_kernel<<<BH_, 256, 0, stream>>>(Qlm, Klm, Kinv);
    k3v_kernel<<<BH_, 256, 0, stream>>>(Qlm, Kbf, Vf, Kinv, MT);
    sv_kernel<<<dim3(BH_, N_ / 128), 256, 0, stream>>>(Qbf, Klm, MT, SVbf);
    gemm_out<<<dim3(H_, 128), 256, 0, stream>>>(SVbf, wprojT, b_proj, Vf, out);
}
